// BiLSTM_CRF_57045755625565
// MI455X (gfx1250) — compile-verified, matrix operand order checked by reading, not run on hardware
//
#include <hip/hip_runtime.h>
#include <hip/hip_bf16.h>
#include <stdint.h>
#include <stddef.h>

#define T_LEN 2048
#define D_IN  2048
#define HD    1024
#define G4    4096      // 4*HD
#define KTAGS 22
#define END_TAG 20
#define START_TAG 21
#define NEGV  -10000.0f

typedef __bf16 v16bf __attribute__((ext_vector_type(16)));
typedef float  v8f   __attribute__((ext_vector_type(8)));

__device__ __forceinline__ unsigned short f32_to_bf16(float f) {
    union { float f; uint32_t u; } in; in.f = f;
    uint32_t u = in.u;
    if ((u & 0x7fffffffu) > 0x7f800000u) return (unsigned short)((u >> 16) | 0x40); // quiet NaN
    uint32_t r = u + 0x7fffu + ((u >> 16) & 1u);                                   // RNE
    return (unsigned short)(r >> 16);
}

__device__ __forceinline__ float bf16_to_f32(unsigned short h) {
    union { uint32_t u; float f; } cv; cv.u = ((uint32_t)h) << 16;
    return cv.f;
}

// ---------------------------------------------------------------- utilities
__global__ void zero_i32_kernel(int* __restrict__ p, int n) {
    int i = blockIdx.x * blockDim.x + threadIdx.x;
    int stride = gridDim.x * blockDim.x;
    for (; i < n; i += stride) p[i] = 0;
}

__global__ void cvt_f32_bf16_kernel(const float* __restrict__ src,
                                    unsigned short* __restrict__ dst, int n) {
    int i = blockIdx.x * blockDim.x + threadIdx.x;
    int stride = gridDim.x * blockDim.x;
    for (; i < n; i += stride) dst[i] = f32_to_bf16(src[i]);
}

// ---------------------------------------------------------------- WMMA GEMM
// C[m,n] = sum_k A[m,k] * B[n,k] + bias1[n] + bias2[n]
// A: [M x K] bf16 row-major (optionally time-reversed rows)
// B: [N x K] bf16 row-major (acts as B^T of a KxN matrix)
// Wave tile: 32(M) x 64(N) -> 2 A-frags x 4 B-frags = 8 WMMAs / 32-K step.
// Block: 256 thr = 8 waves (2x4) -> 64(M) x 256(N). Grid: (N/256, M/64).
__global__ void gemm_bf16_wmma_kernel(const unsigned short* __restrict__ A,
                                      const unsigned short* __restrict__ B,
                                      const float* __restrict__ bias1,
                                      const float* __restrict__ bias2,
                                      float* __restrict__ C,
                                      int M, int N, int Kdim, int reverseA)
{
    const int lane = threadIdx.x & 31;
    const int wave = threadIdx.x >> 5;
    const int wm = wave >> 2;          // 0..1
    const int wn = wave & 3;           // 0..3
    const int m0 = blockIdx.y * 64 + wm * 32;
    const int n0 = blockIdx.x * 256 + wn * 64;
    const int hl  = lane >> 4;         // half-wave: selects K-subgroup
    const int l15 = lane & 15;

    int arow0 = m0 + l15;
    int arow1 = m0 + 16 + l15;
    if (reverseA) { arow0 = M - 1 - arow0; arow1 = M - 1 - arow1; }
    const unsigned short* aptr0 = A + (size_t)arow0 * Kdim;
    const unsigned short* aptr1 = A + (size_t)arow1 * Kdim;
    const unsigned short* bptr[4];
#pragma unroll
    for (int j = 0; j < 4; ++j)
        bptr[j] = B + (size_t)(n0 + j * 16 + l15) * Kdim;

    v8f acc[8];
#pragma unroll
    for (int j = 0; j < 8; ++j)
#pragma unroll
        for (int r = 0; r < 8; ++r) acc[j][r] = 0.0f;

    for (int k0 = 0; k0 < Kdim; k0 += 32) {
        // Prefetch next k-step lines into cache (global_prefetch_b8).
        if (k0 + 64 < Kdim) {
            __builtin_prefetch(aptr0 + k0 + 64, 0, 1);
            __builtin_prefetch(aptr1 + k0 + 64, 0, 1);
#pragma unroll
            for (int j = 0; j < 4; ++j) __builtin_prefetch(bptr[j] + k0 + 64, 0, 1);
        }
        // A fragments: 16-bit A 16x32 layout (ISA 7.12.2)
        v16bf a0, a1;
#pragma unroll
        for (int i = 0; i < 16; ++i) {
            int v = i >> 1, pos = i & 1;
            int k = k0 + ((v < 4) ? 0 : 16) + hl * 8 + (v & 3) * 2 + pos;
            a0[i] = __builtin_bit_cast(__bf16, aptr0[k]);
            a1[i] = __builtin_bit_cast(__bf16, aptr1[k]);
        }
#pragma unroll
        for (int j = 0; j < 4; ++j) {
            v16bf b;
#pragma unroll
            for (int i = 0; i < 16; ++i) {
                int v = i >> 1, pos = i & 1;
                int k = k0 + ((v < 4) ? 0 : 16) + hl * 8 + (v & 3) * 2 + pos;
                b[i] = __builtin_bit_cast(__bf16, bptr[j][k]);
            }
            acc[j] = __builtin_amdgcn_wmma_f32_16x16x32_bf16(
                false, a0, false, b, (short)0, acc[j], false, false);
            acc[4 + j] = __builtin_amdgcn_wmma_f32_16x16x32_bf16(
                false, a1, false, b, (short)0, acc[4 + j], false, false);
        }
    }

    // Epilogue: f32 C/D layout — lane n = lane&15, VGPR r -> m = r + (lane>=16)*8
#pragma unroll
    for (int j = 0; j < 4; ++j) {
        int n = n0 + j * 16 + l15;
        float bsum = bias1[n] + bias2[n];
#pragma unroll
        for (int r = 0; r < 8; ++r) {
            int m = m0 + hl * 8 + r;
            C[(size_t)m * N + n]        = acc[j][r] + bsum;
            C[(size_t)(m + 16) * N + n] = acc[4 + j][r] + bsum;
        }
    }
}

// ---------------------------------------------------------------- persistent LSTM
// 64 workgroups: wg 0..31 forward dir, 32..63 reverse dir; each owns 32 hidden units.
// Whh slab (4 gates x 32 units x 1024) cached in LDS as bf16 (256 KB < 320 KB/WGP).
// Per-timestep cross-WG sync via release/acquire flags in global memory.
// h-vector staged into LDS each step via GLOBAL_LOAD_ASYNC_TO_LDS_B128 (ASYNCcnt).
__global__ void lstm_persistent_kernel(const float* __restrict__ P_f,
                                       const float* __restrict__ P_r,
                                       const float* __restrict__ Whh_f,
                                       const float* __restrict__ Whh_r,
                                       const float* __restrict__ h0,
                                       const float* __restrict__ c0,
                                       float* __restrict__ hseq,
                                       float* __restrict__ hs_out,
                                       int* __restrict__ flags)
{
    extern __shared__ char smem[];
    unsigned short* slab = (unsigned short*)smem;              // 128*1024 bf16 = 256 KB
    float* hprev = (float*)(smem + 128 * 1024 * 2);            // 1024 f32
    float* part  = hprev + HD;                                 // 256 f32

    const int tid = threadIdx.x;
    const int wg  = blockIdx.x;
    const int dir = wg >> 5;
    const int unitBase = (wg & 31) * 32;
    const float* P   = dir ? P_r   : P_f;
    const float* Whh = dir ? Whh_r : Whh_f;

    // Load & convert this WG's Whh slab into LDS once.
    for (int idx = tid; idx < 128 * HD; idx += blockDim.x) {
        int rl = idx >> 10;            // local row 0..127
        int k  = idx & 1023;
        int g = rl >> 5, u = rl & 31;
        int grow = g * HD + unitBase + u;
        slab[idx] = f32_to_bf16(Whh[(size_t)grow * HD + k]);
    }
    __syncthreads();

    const int row   = tid >> 1;        // 0..127
    const int half  = tid & 1;
    const int kbase = half * 512;
    // per-lane LDS byte offset for async h staging (16 B per lane, 4 KB total)
    const uint32_t hprev_lds_off = (uint32_t)(size_t)(hprev + tid * 4);

    float cstate = 0.0f;
    if (tid < 32) cstate = c0[dir * HD + unitBase + tid];

    for (int s = 0; s < T_LEN; ++s) {
        if (s > 0) {
            if (tid == 0) {
                int* f = flags + dir * T_LEN + (s - 1);
                while (__hip_atomic_load(f, __ATOMIC_ACQUIRE,
                                         __HIP_MEMORY_SCOPE_AGENT) < 32)
                    __builtin_amdgcn_s_sleep(1);
            }
            __syncthreads();
        }
        const float* hsrc = (s == 0) ? (h0 + dir * HD)
                                     : (hseq + (size_t)(dir * T_LEN + s - 1) * HD);
        {
            // async copy: LDS[hprev + tid*16B] = MEM[hsrc + tid*16B]  (b128/lane)
            uint64_t gaddr = (uint64_t)(size_t)(hsrc + tid * 4);
            asm volatile("global_load_async_to_lds_b128 %0, %1, off"
                         :: "v"(hprev_lds_off), "v"(gaddr) : "memory");
            asm volatile("s_wait_asynccnt 0x0" ::: "memory");
        }
        __syncthreads();

        // half-row dot product (512 FMAs per thread)
        float acc = 0.0f;
        const unsigned short* wrow = slab + row * HD + kbase;
        const float* hp = hprev + kbase;
#pragma unroll 8
        for (int k = 0; k < 512; ++k)
            acc = fmaf(bf16_to_f32(wrow[k]), hp[k], acc);
        part[tid] = acc;
        __syncthreads();

        if (tid < 32) {
            int u = tid;
            size_t pb = (size_t)s * G4 + unitBase + u;
            float gi = P[pb + 0 * HD] + part[(0 * 32 + u) * 2] + part[(0 * 32 + u) * 2 + 1];
            float gf = P[pb + 1 * HD] + part[(1 * 32 + u) * 2] + part[(1 * 32 + u) * 2 + 1];
            float gg = P[pb + 2 * HD] + part[(2 * 32 + u) * 2] + part[(2 * 32 + u) * 2 + 1];
            float go = P[pb + 3 * HD] + part[(3 * 32 + u) * 2] + part[(3 * 32 + u) * 2 + 1];
            float i = 1.0f / (1.0f + expf(-gi));
            float f = 1.0f / (1.0f + expf(-gf));
            float g = tanhf(gg);
            float o = 1.0f / (1.0f + expf(-go));
            cstate = f * cstate + i * g;
            float h = o * tanhf(cstate);
            hseq[(size_t)(dir * T_LEN + s) * HD + unitBase + u] = h;
            int t = dir ? (T_LEN - 1 - s) : s;
            hs_out[(size_t)t * (2 * HD) + dir * HD + unitBase + u] = h;
        }
        __threadfence();
        __syncthreads();
        if (tid == 0)
            __hip_atomic_fetch_add(flags + dir * T_LEN + s, 1,
                                   __ATOMIC_RELEASE, __HIP_MEMORY_SCOPE_AGENT);
    }
}

// ---------------------------------------------------------------- tag projection
// feats[t,k] = hs[t,:] . W_tag[k,:] + b_tag[k]
__global__ void tag_proj_kernel(const float* __restrict__ hs,
                                const float* __restrict__ Wtag,
                                const float* __restrict__ btag,
                                float* __restrict__ feats)
{
    __shared__ float part[256];
    const int t = blockIdx.x;
    const int tid = threadIdx.x;
    const int k = tid >> 3;    // 0..31 (22 active)
    const int r = tid & 7;
    float acc = 0.0f;
    if (k < KTAGS) {
        const float* hrow = hs + (size_t)t * (2 * HD);
        const float* wrow = Wtag + (size_t)k * (2 * HD);
        int j0 = r * 256;
#pragma unroll 4
        for (int j = j0; j < j0 + 256; ++j) acc = fmaf(hrow[j], wrow[j], acc);
    }
    part[tid] = acc;
    __syncthreads();
    if (r == 0 && k < KTAGS) {
        float s = btag[k];
#pragma unroll
        for (int q = 0; q < 8; ++q) s += part[k * 8 + q];
        feats[t * KTAGS + k] = s;
    }
}

// ---------------------------------------------------------------- Viterbi (single wave32)
__global__ void viterbi_kernel(const float* __restrict__ feats,
                               const float* __restrict__ trans,
                               int* __restrict__ bps,
                               float* __restrict__ out)
{
    __shared__ float tr[KTAGS * KTAGS];
    __shared__ float fvArr[KTAGS];
    const int lane = threadIdx.x;
    for (int i = lane; i < KTAGS * KTAGS; i += 32) tr[i] = trans[i];
    __syncthreads();

    float fv = (lane == START_TAG) ? 0.0f : NEGV;
    for (int t = 0; t < T_LEN; ++t) {
        if (lane < KTAGS) fvArr[lane] = fv;
        __syncthreads();
        if (lane < KTAGS) {
            float best = -3.4e38f; int arg = 0;
            for (int p = 0; p < KTAGS; ++p) {
                float v = tr[lane * KTAGS + p] + fvArr[p];
                if (v > best) { best = v; arg = p; }   // first max wins (matches argmax)
            }
            fv = best + feats[t * KTAGS + lane];
            bps[t * KTAGS + lane] = arg;
        }
        __syncthreads();
    }
    if (lane < KTAGS) fvArr[lane] = fv + tr[END_TAG * KTAGS + lane];
    __syncthreads();
    if (lane == 0) {
        float best = -3.4e38f; int arg = 0;
        for (int p = 0; p < KTAGS; ++p)
            if (fvArr[p] > best) { best = fvArr[p]; arg = p; }
        out[0] = best;
        int cur = arg;
        for (int t = T_LEN - 1; t >= 0; --t) {
            out[1 + t] = (float)cur;          // path[t]
            cur = bps[t * KTAGS + cur];
        }
    }
}

// ---------------------------------------------------------------- launcher
extern "C" void kernel_launch(void* const* d_in, const int* in_sizes, int n_in,
                              void* d_out, int out_size, void* d_ws, size_t ws_size,
                              hipStream_t stream) {
    (void)in_sizes; (void)n_in; (void)out_size; (void)ws_size;

    const float* seq    = (const float*)d_in[0];
    const float* h0     = (const float*)d_in[1];
    const float* c0     = (const float*)d_in[2];
    const float* Wih_f  = (const float*)d_in[3];
    const float* Whh_f  = (const float*)d_in[4];
    const float* bih_f  = (const float*)d_in[5];
    const float* bhh_f  = (const float*)d_in[6];
    const float* Wih_r  = (const float*)d_in[7];
    const float* Whh_r  = (const float*)d_in[8];
    const float* bih_r  = (const float*)d_in[9];
    const float* bhh_r  = (const float*)d_in[10];
    const float* W_tag  = (const float*)d_in[11];
    const float* b_tag  = (const float*)d_in[12];
    const float* transitions = (const float*)d_in[13];
    float* out = (float*)d_out;

    // Workspace carve-up
    char* ws = (char*)d_ws;
    size_t off = 0;
    unsigned short* seqb  = (unsigned short*)(ws + off); off += (size_t)T_LEN * D_IN * 2;   // 8 MB
    unsigned short* wihfb = (unsigned short*)(ws + off); off += (size_t)G4 * D_IN * 2;      // 16 MB
    unsigned short* wihrb = (unsigned short*)(ws + off); off += (size_t)G4 * D_IN * 2;      // 16 MB
    float* P_f    = (float*)(ws + off); off += (size_t)T_LEN * G4 * 4;                      // 32 MB
    float* P_r    = (float*)(ws + off); off += (size_t)T_LEN * G4 * 4;                      // 32 MB
    float* hseq   = (float*)(ws + off); off += (size_t)2 * T_LEN * HD * 4;                  // 16 MB
    float* hs_out = (float*)(ws + off); off += (size_t)T_LEN * 2 * HD * 4;                  // 16 MB
    float* feats  = (float*)(ws + off); off += (size_t)T_LEN * KTAGS * 4;
    int*   bps    = (int*)  (ws + off); off += (size_t)T_LEN * KTAGS * 4;
    int*   flags  = (int*)  (ws + off); off += (size_t)2 * T_LEN * 4;

    // 1. zero sync flags (ws is poisoned by harness)
    zero_i32_kernel<<<16, 256, 0, stream>>>(flags, 2 * T_LEN);

    // 2. f32 -> bf16 conversions
    cvt_f32_bf16_kernel<<<512, 256, 0, stream>>>(seq,   seqb,  T_LEN * D_IN);
    cvt_f32_bf16_kernel<<<1024, 256, 0, stream>>>(Wih_f, wihfb, G4 * D_IN);
    cvt_f32_bf16_kernel<<<1024, 256, 0, stream>>>(Wih_r, wihrb, G4 * D_IN);

    // 3. Input projections via WMMA: P = seq @ Wih^T + (bih + bhh)
    dim3 ggrid(G4 / 256, T_LEN / 64);
    gemm_bf16_wmma_kernel<<<ggrid, 256, 0, stream>>>(seqb, wihfb, bih_f, bhh_f, P_f,
                                                     T_LEN, G4, D_IN, 0);
    gemm_bf16_wmma_kernel<<<ggrid, 256, 0, stream>>>(seqb, wihrb, bih_r, bhh_r, P_r,
                                                     T_LEN, G4, D_IN, 1);

    // 4. Persistent bidirectional LSTM (64 WGs, 256 KB Whh slab in LDS each)
    size_t lstm_lds = (size_t)128 * 1024 * 2 + HD * 4 + 256 * 4;   // 267264 B
    hipFuncSetAttribute((const void*)lstm_persistent_kernel,
                        hipFuncAttributeMaxDynamicSharedMemorySize, (int)lstm_lds);
    lstm_persistent_kernel<<<64, 256, lstm_lds, stream>>>(P_f, P_r, Whh_f, Whh_r,
                                                          h0, c0, hseq, hs_out, flags);

    // 5. Tag projection
    tag_proj_kernel<<<T_LEN, 256, 0, stream>>>(hs_out, W_tag, b_tag, feats);

    // 6. Viterbi DP + backtrace -> d_out = [score, path(2048)]
    viterbi_kernel<<<1, 32, 0, stream>>>(feats, transitions, bps, out);
}
